// RecurrentFormulationNet1_21784074126008
// MI455X (gfx1250) — matrix operand order repaired and checked by reading, not compiled, not run on hardware
//
#include <hip/hip_runtime.h>
#include <hip/hip_bf16.h>

#define N_NODES 50000
#define N_EDGES 800000
#define HID 128
#define NFEAT 4
#define NMESH 8
#define DMID 8
#define EPS_C 1e-5f
#define DT_C 0.1f

typedef __bf16 bf16;
typedef __attribute__((ext_vector_type(16))) __bf16 v16bf;
typedef __attribute__((ext_vector_type(8)))  float  v8f;

// ---------------- degree / normalization ----------------
__global__ void k_init_deg(float* deg, int n) {
  int i = blockIdx.x * blockDim.x + threadIdx.x;
  if (i < n) deg[i] = 1.0f;              // self-loop
}
__global__ void k_count_deg(const int* __restrict__ dst, int e, float* deg) {
  int i = blockIdx.x * blockDim.x + threadIdx.x;
  if (i < e) atomicAdd(&deg[dst[i]], 1.0f);
}
__global__ void k_rsqrt_inplace(float* d, int n) {
  int i = blockIdx.x * blockDim.x + threadIdx.x;
  if (i < n) d[i] = rsqrtf(d[i]);        // deg >= 1 always
}

// ---------------- weight packing into WMMA B-fragment layout ----------------
// Fragment index: (((ct*nchunks + c)*32 + lane)*8 + v) dwords; dword v of lane
// holds W[c*32 + (lane>=16?16:0) + 2v][ct*16 + (lane&15)] (lo) and k+1 (hi).
__global__ void k_pack_w(const float* __restrict__ W, int K, int Hn, int nchunks,
                         int total, bf16* __restrict__ out) {
  int idx = blockIdx.x * blockDim.x + threadIdx.x;
  if (idx >= total) return;
  int v    = idx & 7;
  int lane = (idx >> 3) & 31;
  int rest = idx >> 8;
  int c  = rest % nchunks;
  int ct = rest / nchunks;
  int hi = lane >> 4;
  int n  = ct * 16 + (lane & 15);
  int k0 = c * 32 + hi * 16 + 2 * v;
  float f0 = (k0     < K && n < Hn) ? W[(size_t)k0 * Hn + n]       : 0.0f;
  float f1 = (k0 + 1 < K && n < Hn) ? W[(size_t)(k0 + 1) * Hn + n] : 0.0f;
  out[(size_t)idx * 2]     = (bf16)f0;
  out[(size_t)idx * 2 + 1] = (bf16)f1;
}

// -------- bf16 WMMA GEMM with fused conv-init epilogue --------
// h[N x Hs] = X[N x Kpad] @ W ; agg[r,n] = bias[n] + h[r,n]*dinv[r]^2 (n < Hc)
__global__ void k_matmul(const bf16* __restrict__ X, int Kpad, int nchunks,
                         const bf16* __restrict__ Bf, int nct,
                         float* __restrict__ h, float* __restrict__ agg,
                         const float* __restrict__ dinv, const float* __restrict__ bias,
                         int Hs, int Hc, int nmt) {
  int wave = threadIdx.x >> 5;
  int lane = threadIdx.x & 31;
  int tile = blockIdx.x * (blockDim.x >> 5) + wave;
  if (tile >= nmt * nct) return;         // wave-uniform: EXEC all-ones inside
  int mt = tile / nct;
  int ct = tile - mt * nct;
  int hi = lane >> 4;
  int m  = mt * 16 + (lane & 15);
  const bf16* xrow  = X + (size_t)m * Kpad;
  const bf16* bbase = Bf + ((size_t)ct * nchunks * 32 + lane) * 16;
  v8f acc = {};
  for (int c = 0; c < nchunks; ++c) {
    union { uint4 u[2]; v16bf v; } A, B;
    int k0 = c * 32 + hi * 16;           // A: lanes0-15 K=0-15; lanes16-31 K=16-31
    A.u[0] = *reinterpret_cast<const uint4*>(xrow + k0);
    A.u[1] = *reinterpret_cast<const uint4*>(xrow + k0 + 8);
    const bf16* bp = bbase + (size_t)c * 512;
    B.u[0] = *reinterpret_cast<const uint4*>(bp);
    B.u[1] = *reinterpret_cast<const uint4*>(bp + 8);
    acc = __builtin_amdgcn_wmma_f32_16x16x32_bf16(false, A.v, false, B.v,
                                                  (short)0, acc, false, false);
  }
  int n  = ct * 16 + (lane & 15);
  int r0 = mt * 16 + hi * 8;
  float bn = (n < Hc) ? bias[n] : 0.0f;
#pragma unroll
  for (int v = 0; v < 8; ++v) {
    int r = r0 + v;
    h[(size_t)r * Hs + n] = acc[v];
    if (n < Hc) {
      float dv = dinv[r];
      agg[(size_t)r * Hs + n] = bn + acc[v] * (dv * dv);
    }
  }
}

// ---------------- edge scatter: one thread = one edge x 4 channels ----------------
__global__ void k_scatter4(const float* __restrict__ h, const float* __restrict__ dinv,
                           const int* __restrict__ src, const int* __restrict__ dst,
                           float* __restrict__ agg, long long total, int qshift, int Hs) {
  long long idx = (long long)blockIdx.x * blockDim.x + threadIdx.x;
  if (idx >= total) return;
  int e = (int)(idx >> qshift);
  int g = (int)(idx & ((1 << qshift) - 1));
  int s = src[e], d = dst[e];
  float en = dinv[s] * dinv[d];
  const float4 hv = *reinterpret_cast<const float4*>(h + (size_t)s * Hs + g * 4);
  float* ap = agg + (size_t)d * Hs + g * 4;
  atomicAdd(ap + 0, hv.x * en);
  atomicAdd(ap + 1, hv.y * en);
  atomicAdd(ap + 2, hv.z * en);
  atomicAdd(ap + 3, hv.w * en);
}

// ---------------- instance norm stats over N per channel (Hs=128) ----------------
__global__ void k_stats(const float* __restrict__ x, int n, float* __restrict__ sums) {
  int c    = threadIdx.x & 127;
  int half = threadIdx.x >> 7;
  __shared__ float sh[256], shq[256];
  int rend = blockIdx.x * 128 + 128; if (rend > n) rend = n;
  float s = 0.f, q = 0.f;
  for (int r = blockIdx.x * 128 + half; r < rend; r += 2) {
    float v = x[(size_t)r * 128 + c];
    s += v; q += v * v;
  }
  sh[threadIdx.x] = s; shq[threadIdx.x] = q;
  __syncthreads();
  if (half == 0) {
    atomicAdd(&sums[c],       sh[c]  + sh[c + 128]);
    atomicAdd(&sums[128 + c], shq[c] + shq[c + 128]);
  }
}
__global__ void k_fin_stats(const float* __restrict__ sums, float* __restrict__ ms, int n) {
  int c = threadIdx.x;
  float inv  = 1.0f / (float)n;
  float mean = sums[c] * inv;
  float var  = sums[128 + c] * inv - mean * mean;
  ms[c]       = mean;
  ms[128 + c] = rsqrtf(var + EPS_C);
}

// ---- activation -> padded bf16, 4 channels per thread (mode 0 plain,1 relu,2 inorm+relu)
__global__ void k_convert4(const float* __restrict__ x, const float* __restrict__ ms,
                           bf16* __restrict__ out, int n, int Hc, int Hs, int Kpad, int mode) {
  int q   = Kpad >> 2;
  int idx = blockIdx.x * blockDim.x + threadIdx.x;
  if (idx >= n * q) return;
  int i  = idx / q;
  int c4 = (idx - i * q) * 4;
  union { bf16 b[4]; uint2 u; } o;
  if (c4 < Hc) {                         // Hc, Kpad multiples of 4: fully in-range
    float4 v = *reinterpret_cast<const float4*>(x + (size_t)i * Hs + c4);
    float f[4] = { v.x, v.y, v.z, v.w };
#pragma unroll
    for (int j = 0; j < 4; ++j) {
      float t = f[j];
      if (mode == 2) t = (t - ms[c4 + j]) * ms[128 + c4 + j];
      if (mode >= 1) t = fmaxf(t, 0.0f);
      o.b[j] = (bf16)t;
    }
  } else {
    o.b[0] = o.b[1] = o.b[2] = o.b[3] = (bf16)0.0f;
  }
  *reinterpret_cast<uint2*>(out + (size_t)i * Kpad + c4) = o.u;
}

__global__ void k_xcat4(const float* __restrict__ Fp, const float* __restrict__ mesh,
                        bf16* __restrict__ out, int n) {
  int idx = blockIdx.x * blockDim.x + threadIdx.x;   // n * 40
  if (idx >= n * 40) return;
  int i  = idx / 40;
  int c4 = (idx - i * 40) * 4;
  union { bf16 b[4]; uint2 u; } o;
  float4 v = { 0.f, 0.f, 0.f, 0.f };
  if (c4 < NFEAT)                 v = *reinterpret_cast<const float4*>(Fp + (size_t)i * NFEAT);
  else if (c4 < NFEAT + HID)      v = *reinterpret_cast<const float4*>(mesh + (size_t)i * HID + (c4 - NFEAT));
  o.b[0] = (bf16)v.x; o.b[1] = (bf16)v.y; o.b[2] = (bf16)v.z; o.b[3] = (bf16)v.w;
  *reinterpret_cast<uint2*>(out + (size_t)i * 160 + c4) = o.u;
}

__global__ void k_tanh(const float* __restrict__ in, float* __restrict__ out, size_t n) {
  size_t i = (size_t)blockIdx.x * blockDim.x + threadIdx.x;
  if (i < n) out[i] = tanhf(in[i]);
}
__global__ void k_copy(const float* __restrict__ a, float* __restrict__ b, int n) {
  int i = blockIdx.x * blockDim.x + threadIdx.x;
  if (i < n) b[i] = a[i];
}

__global__ void k_step(const float* __restrict__ agg, float* __restrict__ Fp,
                       float* __restrict__ Fs, float* __restrict__ Fds,
                       int n, int n_time, int t) {
  int idx = blockIdx.x * blockDim.x + threadIdx.x;
  if (idx >= n * NFEAT) return;
  int i = idx >> 2;
  int c = idx & 3;
  float fd = tanhf(agg[(size_t)i * 16 + c]);
  float fn = tanhf(Fp[idx] + fd * DT_C);
  size_t o = ((size_t)i * n_time + t) * NFEAT + c;
  Fs[o]  = fn;
  Fds[o] = fd;
  Fp[idx] = fn;
}

// =====================================================================
extern "C" void kernel_launch(void* const* d_in, const int* in_sizes, int n_in,
                              void* d_out, int out_size, void* d_ws, size_t ws_size,
                              hipStream_t stream) {
  const int N = N_NODES, E = N_EDGES;
  const float* F0   = (const float*)d_in[0];
  const float* mfld = (const float*)d_in[1];
  const int*   ei   = (const int*)d_in[2];
  const float* mW0 = (const float*)d_in[4];
  const float* mb0 = (const float*)d_in[5];
  const float* mWm = (const float*)d_in[6];
  const float* mbm = (const float*)d_in[7];
  const float* mWl = (const float*)d_in[8];
  const float* mbl = (const float*)d_in[9];
  const float* dW0 = (const float*)d_in[10];
  const float* db0 = (const float*)d_in[11];
  const float* dWm = (const float*)d_in[12];
  const float* dbm = (const float*)d_in[13];
  const float* dWl = (const float*)d_in[14];
  const float* dbl = (const float*)d_in[15];
  const int* srcp = ei;
  const int* dstp = ei + E;

  int n_time = out_size / (2 * N * NFEAT);
  if (n_time < 1) n_time = 1;

  char* wsp = (char*)d_ws;
  auto alloc = [&](size_t bytes) {
    char* p = wsp;
    wsp += (bytes + 255) & ~(size_t)255;
    return p;
  };
  float* dinv = (float*)alloc((size_t)N * 4);
  float* sums = (float*)alloc(1024);
  float* ms   = (float*)alloc(1024);
  float* h    = (float*)alloc((size_t)N * HID * 4);
  float* agg  = (float*)alloc((size_t)N * HID * 4);
  bf16*  xb   = (bf16*) alloc((size_t)N * 160 * 2);
  float* mesh = (float*)alloc((size_t)N * HID * 4);
  float* Fp   = (float*)alloc((size_t)N * NFEAT * 4);
  // weight fragments: nct*nchunks*1024 bytes each
  bf16* f_mW0 = (bf16*)alloc((size_t)8 * 1 * 1024);
  bf16* f_mWm = (bf16*)alloc((size_t)DMID * 8 * 4 * 1024);
  bf16* f_mWl = (bf16*)alloc((size_t)8 * 4 * 1024);
  bf16* f_dW0 = (bf16*)alloc((size_t)8 * 5 * 1024);
  bf16* f_dWm = (bf16*)alloc((size_t)DMID * 8 * 4 * 1024);
  bf16* f_dWl = (bf16*)alloc((size_t)1 * 4 * 1024);

  auto pack = [&](const float* W, int K, int Hn, int nchunks, int nct, bf16* out) {
    int total = nct * nchunks * 256;
    k_pack_w<<<(total + 255) / 256, 256, 0, stream>>>(W, K, Hn, nchunks, total, out);
  };
  const size_t FRAG_MID = (size_t)8 * 4 * 512;   // bf16 elements per 128x128 matrix
  pack(mW0, NMESH, HID, 1, 8, f_mW0);
  for (int i = 0; i < DMID; ++i)
    pack(mWm + (size_t)i * HID * HID, HID, HID, 4, 8, f_mWm + (size_t)i * FRAG_MID);
  pack(mWl, HID, HID, 4, 8, f_mWl);
  pack(dW0, NFEAT + HID, HID, 5, 8, f_dW0);
  for (int i = 0; i < DMID; ++i)
    pack(dWm + (size_t)i * HID * HID, HID, HID, 4, 8, f_dWm + (size_t)i * FRAG_MID);
  pack(dWl, HID, NFEAT, 4, 1, f_dWl);

  // degrees + symmetric norm
  k_init_deg<<<(N + 255) / 256, 256, 0, stream>>>(dinv, N);
  k_count_deg<<<(E + 255) / 256, 256, 0, stream>>>(dstp, E, dinv);
  k_rsqrt_inplace<<<(N + 255) / 256, 256, 0, stream>>>(dinv, N);

  auto conv = [&](const bf16* X, int Kpad, int nchunks, const bf16* Bf,
                  const float* bias, int Hc, int Hs) {
    int nct = Hs / 16;
    int nmt = N / 16;                    // 3125, exact
    int tiles = nmt * nct;
    k_matmul<<<(tiles + 7) / 8, 256, 0, stream>>>(X, Kpad, nchunks, Bf, nct,
                                                  h, agg, dinv, bias, Hs, Hc, nmt);
    int qshift = (Hc == 128) ? 5 : 0;    // (Hc/4) groups per edge
    long long total = (long long)E << qshift;
    k_scatter4<<<(unsigned)((total + 255) / 256), 256, 0, stream>>>(
        h, dinv, srcp, dstp, agg, total, qshift, Hs);
  };

  auto blockrun = [&](const bf16* x0, int Kpad0, int nch0,
                      const bf16* W0f, const float* b0,
                      const bf16* Wmf, const float* bm,
                      const bf16* Wlf, const float* bl, int HcL, int HsL) {
    conv(x0, Kpad0, nch0, W0f, b0, HID, HID);
    hipMemsetAsync(sums, 0, 1024, stream);
    k_stats<<<(N + 127) / 128, 256, 0, stream>>>(agg, N, sums);
    k_fin_stats<<<1, 128, 0, stream>>>(sums, ms, N);
    k_convert4<<<(N * 32 + 255) / 256, 256, 0, stream>>>(agg, ms, xb, N, HID, HID, HID, 2);
    for (int i = 0; i < DMID; ++i) {
      conv(xb, HID, 4, Wmf + (size_t)i * FRAG_MID, bm + (size_t)i * HID, HID, HID);
      k_convert4<<<(N * 32 + 255) / 256, 256, 0, stream>>>(
          agg, ms, xb, N, HID, HID, HID, (i < DMID - 1) ? 1 : 0);
    }
    conv(xb, HID, 4, Wlf, bl, HcL, HsL);
  };

  // ---- mesh branch (computed once) ----
  k_convert4<<<(N * 8 + 255) / 256, 256, 0, stream>>>(mfld, nullptr, xb, N, NMESH, NMESH, 32, 0);
  blockrun(xb, 32, 1, f_mW0, mb0, f_mWm, mbm, f_mWl, mbl, HID, HID);
  k_tanh<<<((unsigned)(((size_t)N * HID + 255) / 256)), 256, 0, stream>>>(agg, mesh, (size_t)N * HID);

  // ---- time stepping ----
  k_copy<<<(N * NFEAT + 255) / 256, 256, 0, stream>>>(F0, Fp, N * NFEAT);
  float* Fs  = (float*)d_out;
  float* Fds = Fs + (size_t)N * n_time * NFEAT;
  for (int t = 0; t < n_time; ++t) {
    k_xcat4<<<(N * 40 + 255) / 256, 256, 0, stream>>>(Fp, mesh, xb, N);
    blockrun(xb, 160, 5, f_dW0, db0, f_dWm, dbm, f_dWl, dbl, NFEAT, 16);
    k_step<<<(N * NFEAT + 255) / 256, 256, 0, stream>>>(agg, Fp, Fs, Fds, N, n_time, t);
  }
}
